// EncoderLayer_14121852469953
// MI455X (gfx1250) — matrix operand order repaired and checked by reading, not compiled, not run on hardware
//
#include <hip/hip_runtime.h>
#include <hip/hip_bf16.h>
#include <cstdint>
#include <cstddef>

#define D_MODEL  1024
#define N_HEADS  16
#define HEAD_DIM 64
#define D_FF     4096
#define BATCH    2
#define SEQ      2048
#define NTOK     (BATCH * SEQ)   // 4096

typedef __attribute__((ext_vector_type(16))) _Float16 v16h;
typedef __attribute__((ext_vector_type(8)))  _Float16 v8h;
typedef __attribute__((ext_vector_type(8)))  float    v8f;

// ---------------------------------------------------------------------------
// WMMA fragment helpers (CDNA5 layouts, wave32)
// ---------------------------------------------------------------------------
static __device__ __forceinline__ v16h cat8(v8h lo, v8h hi) {
  return __builtin_shufflevector(lo, hi, 0,1,2,3,4,5,6,7,8,9,10,11,12,13,14,15);
}

// A-matrix 16x32 f16 fragment from row-major A (row stride lda, elements).
// lane<16: row = lane,    K = k0+{0..15}
// lane>=16: row = lane-16, K = k0+{16..31}
static __device__ __forceinline__ v16h load_a_frag(const _Float16* A, int lda,
                                                   int m0, int k0, int lane) {
  const _Float16* p = A + (size_t)(m0 + (lane & 15)) * lda
                        + (size_t)(k0 + ((lane >> 4) << 4));
  v8h lo = *(const v8h*)p;
  v8h hi = *(const v8h*)(p + 8);
  return cat8(lo, hi);
}

// B-matrix 32x16 f16 fragment where B[k][n] == BT[n][k], BT row-major (row
// stride ldb). lane<16: col n = lane, K = k0..k0+15; lane>=16: K = k0+16..31.
static __device__ __forceinline__ v16h load_b_frag(const _Float16* BT, int ldb,
                                                   int n0, int k0, int lane) {
  return *(const v16h*)(BT + (size_t)(n0 + (lane & 15)) * ldb
                           + (size_t)(k0 + ((lane >> 4) << 4)));
}

static __device__ __forceinline__ v8f wmma_f16f32(v16h a, v16h b, v8f c) {
  return __builtin_amdgcn_wmma_f32_16x16x32_f16(false, a, false, b,
                                                (short)0, c, false, false);
}

// ---------------------------------------------------------------------------
// CDNA5 async HBM -> LDS copy (ASYNCcnt-tracked), 16 bytes per lane.
// ---------------------------------------------------------------------------
static __device__ __forceinline__ void async_copy_b128(uint32_t lds_off,
                                                       const void* gptr) {
  asm volatile("global_load_async_to_lds_b128 %0, %1, off"
               :: "v"(lds_off), "v"((unsigned long long)(uintptr_t)gptr)
               : "memory");
}
static __device__ __forceinline__ void wait_async_le(int n) {
  if (n == 0)      asm volatile("s_wait_asynccnt 0x0" ::: "memory");
  else if (n == 6) asm volatile("s_wait_asynccnt 0x6" ::: "memory");
  else             asm volatile("s_wait_asynccnt 0x8" ::: "memory");
}

// ---------------------------------------------------------------------------
// WMMA GEMM with double-buffered async LDS staging, 64-wide K stages.
//   C[M,N] = A[M,K] * BT[N,K]^T + bias, epilogue by MODE:
//   MODE 0: f16 row-major   MODE 1: f16 row-major + ReLU
//   MODE 2: f32 row-major   MODE 3: f16 -> [B,H,S,Dh]   MODE 4: f16 -> [B,H,Dh,S]
// Block = 256 threads (8 waves, 2x4), block tile 64(M) x 128(N).
// ---------------------------------------------------------------------------
template<int MODE>
static __device__ __forceinline__ void store_tile(void* Cout, const float* bias,
                                                  v8f c, int m0, int n0, int N,
                                                  int lane) {
  const int col   = n0 + (lane & 15);
  const int rbase = m0 + ((lane >> 4) << 3);
  const float bv  = bias ? bias[col] : 0.0f;
#pragma unroll
  for (int r = 0; r < 8; ++r) {
    const int row = rbase + r;
    float v = c[r] + bv;
    if constexpr (MODE == 1) v = v > 0.0f ? v : 0.0f;
    if constexpr (MODE == 0 || MODE == 1) {
      ((_Float16*)Cout)[(size_t)row * N + col] = (_Float16)v;
    } else if constexpr (MODE == 2) {
      ((float*)Cout)[(size_t)row * N + col] = v;
    } else if constexpr (MODE == 3) {          // Q/K: [B, H, S, Dh]
      const int b = row >> 11, s = row & (SEQ - 1);
      const int h = col >> 6,  d = col & (HEAD_DIM - 1);
      ((_Float16*)Cout)[((((size_t)b * N_HEADS + h) * SEQ) + s) * HEAD_DIM + d] =
          (_Float16)v;
    } else {                                   // V transposed: [B, H, Dh, S]
      const int b = row >> 11, s = row & (SEQ - 1);
      const int h = col >> 6,  d = col & (HEAD_DIM - 1);
      ((_Float16*)Cout)[((((size_t)b * N_HEADS + h) * HEAD_DIM) + d) * SEQ + s] =
          (_Float16)v;
    }
  }
}

template<int MODE>
__global__ __launch_bounds__(256)
void gemm_wmma_kernel(const _Float16* __restrict__ A,
                      const _Float16* __restrict__ BT,
                      const float* __restrict__ bias,
                      void* __restrict__ Cout, int M, int N, int K) {
  __shared__ __align__(32) _Float16 Alds[2][64][64];    // 16 KB
  __shared__ __align__(32) _Float16 Blds[2][128][64];   // 32 KB
  const int lane = threadIdx.x & 31;
  const int w    = threadIdx.x >> 5;       // 0..7
  const int wm   = w & 1, wn = w >> 1;     // 2 x 4 wave grid
  const int m0b  = blockIdx.x * 64;
  const int n0b  = blockIdx.y * 128;
  const int t    = threadIdx.x;

  // Stage one 64-wide K tile: A 2 chunks/thread + B 4 chunks/thread = 6 asyncs
  auto issue_tile = [&](int kt, int buf) {
#pragma unroll
    for (int i = 0; i < 2; ++i) {
      const int c = t + i * 256;                     // 64 rows x 8 segs
      const int row = c >> 3, seg = c & 7;
      async_copy_b128((uint32_t)(uintptr_t)&Alds[buf][row][seg * 8],
                      A + (size_t)(m0b + row) * K + kt + seg * 8);
    }
#pragma unroll
    for (int i = 0; i < 4; ++i) {
      const int c = t + i * 256;                     // 128 rows x 8 segs
      const int row = c >> 3, seg = c & 7;
      async_copy_b128((uint32_t)(uintptr_t)&Blds[buf][row][seg * 8],
                      BT + (size_t)(n0b + row) * K + kt + seg * 8);
    }
  };

  const int NT = K >> 6;
  issue_tile(0, 0);
  issue_tile(64, 1);

  v8f c00 = {}, c01 = {}, c10 = {}, c11 = {};
  for (int i = 0; i < NT; ++i) {
    wait_async_le(i == NT - 1 ? 0 : 6);   // this stage's tile has landed
    __syncthreads();                      // publish tile block-wide
    const int buf = i & 1;
    const _Float16* Ab = &Alds[buf][0][0];
    const _Float16* Bb = &Blds[buf][0][0];
#pragma unroll
    for (int kk = 0; kk < 64; kk += 32) {
      v16h a0 = load_a_frag(Ab, 64, wm * 32,      kk, lane);
      v16h a1 = load_a_frag(Ab, 64, wm * 32 + 16, kk, lane);
      v16h b0 = load_b_frag(Bb, 64, wn * 32,      kk, lane);
      v16h b1 = load_b_frag(Bb, 64, wn * 32 + 16, kk, lane);
      c00 = wmma_f16f32(a0, b0, c00);
      c01 = wmma_f16f32(a0, b1, c01);
      c10 = wmma_f16f32(a1, b0, c10);
      c11 = wmma_f16f32(a1, b1, c11);
    }
    __syncthreads();                      // all reads done before buffer reuse
    if (i + 2 < NT) issue_tile((i + 2) * 64, buf);
  }

  store_tile<MODE>(Cout, bias, c00, m0b + wm * 32,      n0b + wn * 32,      N, lane);
  store_tile<MODE>(Cout, bias, c01, m0b + wm * 32,      n0b + wn * 32 + 16, N, lane);
  store_tile<MODE>(Cout, bias, c10, m0b + wm * 32 + 16, n0b + wn * 32,      N, lane);
  store_tile<MODE>(Cout, bias, c11, m0b + wm * 32 + 16, n0b + wn * 32 + 16, N, lane);
}

// ---------------------------------------------------------------------------
// Flash attention, 64-key chunks async-staged and shared by all 4 waves.
// Q,K: [B,H,S,64] f16; Vt: [B,H,64,S] f16; ctx out: [B,S,D_MODEL] f16.
// ---------------------------------------------------------------------------
__global__ __launch_bounds__(128)
void flash_attn_kernel(const _Float16* __restrict__ Q,
                       const _Float16* __restrict__ Kt,
                       const _Float16* __restrict__ Vt,
                       const int* __restrict__ mask,
                       _Float16* __restrict__ ctx) {
  __shared__ __align__(32) _Float16 Klds[2][64][64];   // 16 KB (keys x d)
  __shared__ __align__(32) _Float16 Vlds[2][64][64];   // 16 KB (d x keys)
  __shared__ __align__(32) _Float16 plds[4][16][64];   // 8 KB  (per-wave P tile)
  const int lane = threadIdx.x & 31;
  const int w    = threadIdx.x >> 5;                 // 0..3
  const int t    = threadIdx.x;
  const int bh   = blockIdx.y;                       // 0..B*H-1
  const int b    = bh >> 4;
  const int head = bh & (N_HEADS - 1);
  const int q0   = (blockIdx.x * 4 + w) * 16;        // query block start

  const _Float16* Qp = Q  + (size_t)bh * SEQ * HEAD_DIM;
  const _Float16* Kp = Kt + (size_t)bh * SEQ * HEAD_DIM;
  const _Float16* Vp = Vt + (size_t)bh * HEAD_DIM * SEQ;
  const int*      mp = mask + (size_t)b * SEQ;

  // Stage one 64-key chunk of K (64x64) and Vt (64x64): 8 asyncs per thread.
  auto issue_kv = [&](int jc, int buf) {
#pragma unroll
    for (int i = 0; i < 4; ++i) {
      const int c = t + i * 128;                     // 64 rows x 8 segs
      const int row = c >> 3, seg = c & 7;
      async_copy_b128((uint32_t)(uintptr_t)&Klds[buf][row][seg * 8],
                      Kp + (size_t)(jc + row) * HEAD_DIM + seg * 8);
    }
#pragma unroll
    for (int i = 0; i < 4; ++i) {
      const int c = t + i * 128;                     // 64 rows x 8 segs
      const int row = c >> 3, seg = c & 7;
      async_copy_b128((uint32_t)(uintptr_t)&Vlds[buf][row][seg * 8],
                      Vp + (size_t)row * SEQ + jc + seg * 8);
    }
  };

  // Q A-fragments (held in registers for the whole key loop)
  const v16h qa0 = load_a_frag(Qp, HEAD_DIM, q0, 0,  lane);
  const v16h qa1 = load_a_frag(Qp, HEAD_DIM, q0, 32, lane);

  issue_kv(0, 0);
  issue_kv(64, 1);

  v8f o0 = {}, o1 = {}, o2 = {}, o3 = {};
  float mrow[8], lrow[8];
#pragma unroll
  for (int r = 0; r < 8; ++r) { mrow[r] = -1e30f; lrow[r] = 0.0f; }

  const int colw  = lane & 15;
  const int rbase = (lane >> 4) << 3;
  const int NCH   = SEQ / 64;   // 32 chunks

  for (int ic = 0; ic < NCH; ++ic) {
    const int jc  = ic * 64;
    const int buf = ic & 1;
    wait_async_le(ic == NCH - 1 ? 0 : 8);
    __syncthreads();                                  // K/V chunk visible

    // ---- scores: four 16x16 tiles (keys jc+16*jt) from LDS ----
    const _Float16* Kb = &Klds[buf][0][0];
    v8f sc[4];
#pragma unroll
    for (int jt = 0; jt < 4; ++jt) {
      v8f s = {};
      v16h kb0 = load_b_frag(Kb, 64, jt * 16, 0,  lane);
      v16h kb1 = load_b_frag(Kb, 64, jt * 16, 32, lane);
      s = wmma_f16f32(qa0, kb0, s);
      s = wmma_f16f32(qa1, kb1, s);
      sc[jt] = s;
    }
    // ---- scale + mask (mask indexed by key) ----
    bool kv[4];
#pragma unroll
    for (int jt = 0; jt < 4; ++jt) kv[jt] = mp[jc + jt * 16 + colw] != 0;
#pragma unroll
    for (int jt = 0; jt < 4; ++jt)
#pragma unroll
      for (int r = 0; r < 8; ++r) {
        const float a = sc[jt][r] * 0.125f;   // 1/sqrt(64)
        sc[jt][r] = kv[jt] ? a : -1e9f;
      }
    // ---- online softmax: row max + rescale ----
    float alpha[8];
#pragma unroll
    for (int r = 0; r < 8; ++r) {
      float mx = fmaxf(fmaxf(sc[0][r], sc[1][r]), fmaxf(sc[2][r], sc[3][r]));
#pragma unroll
      for (int off = 8; off >= 1; off >>= 1)
        mx = fmaxf(mx, __shfl_xor(mx, off, 32));   // 16-lane column groups
      const float mnew = fmaxf(mrow[r], mx);
      alpha[r] = __expf(mrow[r] - mnew);
      mrow[r]  = mnew;
    }
#pragma unroll
    for (int r = 0; r < 8; ++r) {
      float ps = 0.0f;
#pragma unroll
      for (int jt = 0; jt < 4; ++jt) {
        const float p = __expf(sc[jt][r] - mrow[r]);
        sc[jt][r] = p;
        ps += p;
      }
#pragma unroll
      for (int off = 8; off >= 1; off >>= 1) ps += __shfl_xor(ps, off, 32);
      lrow[r] = lrow[r] * alpha[r] + ps;
      o0[r] *= alpha[r]; o1[r] *= alpha[r]; o2[r] *= alpha[r]; o3[r] *= alpha[r];
    }
    // ---- C-layout -> A-layout via per-wave LDS tile (WAR already fenced by
    //      previous end-of-chunk barrier + this chunk's top barrier) ----
#pragma unroll
    for (int jt = 0; jt < 4; ++jt)
#pragma unroll
      for (int r = 0; r < 8; ++r)
        plds[w][rbase + r][jt * 16 + colw] = (_Float16)sc[jt][r];
    __syncthreads();   // DS write->read ordering for the P tile
    const v16h pa0 = load_a_frag(&plds[w][0][0], 64, 0, 0,  lane);
    const v16h pa1 = load_a_frag(&plds[w][0][0], 64, 0, 32, lane);
    // ---- O += P(16x64) x V(64x16) per 16-wide d tile, V from LDS ----
    const _Float16* Vb = &Vlds[buf][0][0];
    {
      v16h va = load_b_frag(Vb, 64, 0, 0, lane), vbq = load_b_frag(Vb, 64, 0, 32, lane);
      o0 = wmma_f16f32(pa0, va, o0); o0 = wmma_f16f32(pa1, vbq, o0);
    }
    {
      v16h va = load_b_frag(Vb, 64, 16, 0, lane), vbq = load_b_frag(Vb, 64, 16, 32, lane);
      o1 = wmma_f16f32(pa0, va, o1); o1 = wmma_f16f32(pa1, vbq, o1);
    }
    {
      v16h va = load_b_frag(Vb, 64, 32, 0, lane), vbq = load_b_frag(Vb, 64, 32, 32, lane);
      o2 = wmma_f16f32(pa0, va, o2); o2 = wmma_f16f32(pa1, vbq, o2);
    }
    {
      v16h va = load_b_frag(Vb, 64, 48, 0, lane), vbq = load_b_frag(Vb, 64, 48, 32, lane);
      o3 = wmma_f16f32(pa0, va, o3); o3 = wmma_f16f32(pa1, vbq, o3);
    }

    __syncthreads();                         // all K/V/P reads done
    if (ic + 2 < NCH) issue_kv(jc + 128, buf);
  }

  // ---- normalize and write ctx [B,S,D_MODEL] (merged heads) ----
#pragma unroll
  for (int r = 0; r < 8; ++r) {
    const float inv = 1.0f / lrow[r];
    const int row = q0 + rbase + r;
    const size_t base = ((size_t)b * SEQ + row) * D_MODEL + head * HEAD_DIM;
    ctx[base + colw]      = (_Float16)(o0[r] * inv);
    ctx[base + 16 + colw] = (_Float16)(o1[r] * inv);
    ctx[base + 32 + colw] = (_Float16)(o2[r] * inv);
    ctx[base + 48 + colw] = (_Float16)(o3[r] * inv);
  }
}

// ---------------------------------------------------------------------------
// Residual + LayerNorm over rows of length D_MODEL (also emits f16 copy)
// ---------------------------------------------------------------------------
static __device__ __forceinline__ float block_sum(float v, float* sh) {
#pragma unroll
  for (int off = 16; off >= 1; off >>= 1) v += __shfl_xor(v, off, 32);
  const int lane = threadIdx.x & 31, wv = threadIdx.x >> 5;
  __syncthreads();
  if (lane == 0) sh[wv] = v;
  __syncthreads();
  float tsum = 0.0f;
#pragma unroll
  for (int i = 0; i < 8; ++i) tsum += sh[i];
  return tsum;
}

__global__ __launch_bounds__(256)
void add_ln_kernel(const float* __restrict__ X, const float* __restrict__ Y,
                   const float* __restrict__ g, const float* __restrict__ bt,
                   float* __restrict__ out32, _Float16* __restrict__ out16) {
  __shared__ float sh[8];
  const int row = blockIdx.x;
  const float* xp = X + (size_t)row * D_MODEL;
  const float* yp = Y + (size_t)row * D_MODEL;
  float vals[4], s = 0.0f;
#pragma unroll
  for (int i = 0; i < 4; ++i) {
    const int c = threadIdx.x + i * 256;
    vals[i] = xp[c] + yp[c];
    s += vals[i];
  }
  const float mu = block_sum(s, sh) * (1.0f / D_MODEL);
  float vs = 0.0f;
#pragma unroll
  for (int i = 0; i < 4; ++i) { const float d = vals[i] - mu; vs += d * d; }
  const float rstd = rsqrtf(block_sum(vs, sh) * (1.0f / D_MODEL) + 1e-5f);
#pragma unroll
  for (int i = 0; i < 4; ++i) {
    const int c = threadIdx.x + i * 256;
    const float o = (vals[i] - mu) * rstd * g[c] + bt[c];
    out32[(size_t)row * D_MODEL + c] = o;
    if (out16) out16[(size_t)row * D_MODEL + c] = (_Float16)o;
  }
}

// ---------------------------------------------------------------------------
// Conversion kernels
// ---------------------------------------------------------------------------
__global__ __launch_bounds__(256)
void cast_f32_to_f16_kernel(const float* __restrict__ in,
                            _Float16* __restrict__ out, int n) {
  const int i = blockIdx.x * 256 + threadIdx.x;
  if (i < n) out[i] = (_Float16)in[i];
}

// in[R][C] f32 -> out[C][R] f16
__global__ __launch_bounds__(256)
void transpose_f32_to_f16_kernel(const float* __restrict__ in,
                                 _Float16* __restrict__ out, int R, int C) {
  const int idx = blockIdx.x * 256 + threadIdx.x;
  if (idx >= R * C) return;
  const int r = idx % R;
  const int c = idx / R;
  out[idx] = (_Float16)in[(size_t)r * C + c];
}

// ---------------------------------------------------------------------------
// Host-side orchestration
// ---------------------------------------------------------------------------
extern "C" void kernel_launch(void* const* d_in, const int* in_sizes, int n_in,
                              void* d_out, int out_size, void* d_ws, size_t ws_size,
                              hipStream_t stream) {
  (void)in_sizes; (void)n_in; (void)out_size; (void)ws_size;
  const float* x    = (const float*)d_in[0];
  const int*   mask = (const int*)  d_in[1];
  const float* Wq = (const float*)d_in[2];  const float* bq = (const float*)d_in[3];
  const float* Wk = (const float*)d_in[4];  const float* bk = (const float*)d_in[5];
  const float* Wv = (const float*)d_in[6];  const float* bv = (const float*)d_in[7];
  const float* Wo = (const float*)d_in[8];  const float* bo = (const float*)d_in[9];
  const float* W1 = (const float*)d_in[10]; const float* b1 = (const float*)d_in[11];
  const float* W2 = (const float*)d_in[12]; const float* b2 = (const float*)d_in[13];
  const float* g1 = (const float*)d_in[14]; const float* be1 = (const float*)d_in[15];
  const float* g2 = (const float*)d_in[16]; const float* be2 = (const float*)d_in[17];
  float* out = (float*)d_out;

  char* ws = (char*)d_ws;
  const size_t MB = 1u << 20;
  _Float16* xh   = (_Float16*)(ws + 0 * MB);     // [4096,1024]
  _Float16* WqT  = (_Float16*)(ws + 8 * MB);     // [1024,1024]
  _Float16* WkT  = (_Float16*)(ws + 10 * MB);
  _Float16* WvT  = (_Float16*)(ws + 12 * MB);
  _Float16* WoT  = (_Float16*)(ws + 14 * MB);
  _Float16* W1T  = (_Float16*)(ws + 16 * MB);    // [4096,1024]
  _Float16* W2T  = (_Float16*)(ws + 24 * MB);    // [1024,4096]
  _Float16* Qb   = (_Float16*)(ws + 32 * MB);    // [B,H,S,64]
  _Float16* Kb   = (_Float16*)(ws + 40 * MB);    // [B,H,S,64]
  _Float16* Vtb  = (_Float16*)(ws + 48 * MB);    // [B,H,64,S]
  _Float16* ctx  = (_Float16*)(ws + 56 * MB);    // [4096,1024]
  float*    attn = (float*)   (ws + 64 * MB);    // [4096,1024]
  float*    x1   = (float*)   (ws + 80 * MB);    // [4096,1024]
  _Float16* x1h  = (_Float16*)(ws + 96 * MB);    // [4096,1024]
  _Float16* hbuf = (_Float16*)(ws + 104 * MB);   // [4096,4096]
  float*    ff   = (float*)   (ws + 136 * MB);   // [4096,1024]

  // 1) Precision conversion + weight transposition
  cast_f32_to_f16_kernel<<<(NTOK * D_MODEL) / 256, 256, 0, stream>>>(x, xh, NTOK * D_MODEL);
  transpose_f32_to_f16_kernel<<<(D_MODEL * D_MODEL) / 256, 256, 0, stream>>>(Wq, WqT, D_MODEL, D_MODEL);
  transpose_f32_to_f16_kernel<<<(D_MODEL * D_MODEL) / 256, 256, 0, stream>>>(Wk, WkT, D_MODEL, D_MODEL);
  transpose_f32_to_f16_kernel<<<(D_MODEL * D_MODEL) / 256, 256, 0, stream>>>(Wv, WvT, D_MODEL, D_MODEL);
  transpose_f32_to_f16_kernel<<<(D_MODEL * D_MODEL) / 256, 256, 0, stream>>>(Wo, WoT, D_MODEL, D_MODEL);
  transpose_f32_to_f16_kernel<<<(D_MODEL * D_FF) / 256, 256, 0, stream>>>(W1, W1T, D_MODEL, D_FF);
  transpose_f32_to_f16_kernel<<<(D_FF * D_MODEL) / 256, 256, 0, stream>>>(W2, W2T, D_FF, D_MODEL);

  // 2) QKV projections (WMMA), Q/K -> [B,H,S,64], V -> [B,H,64,S]
  {
    dim3 grid(NTOK / 64, D_MODEL / 128);
    gemm_wmma_kernel<3><<<grid, 256, 0, stream>>>(xh, WqT, bq, Qb,  NTOK, D_MODEL, D_MODEL);
    gemm_wmma_kernel<3><<<grid, 256, 0, stream>>>(xh, WkT, bk, Kb,  NTOK, D_MODEL, D_MODEL);
    gemm_wmma_kernel<4><<<grid, 256, 0, stream>>>(xh, WvT, bv, Vtb, NTOK, D_MODEL, D_MODEL);
  }

  // 3) Flash attention -> ctx [B,S,D_MODEL] f16
  {
    dim3 grid(SEQ / 64, BATCH * N_HEADS);   // 4 waves/block * 16 rows
    flash_attn_kernel<<<grid, 128, 0, stream>>>(Qb, Kb, Vtb, mask, ctx);
  }

  // 4) Output projection -> attn f32
  {
    dim3 grid(NTOK / 64, D_MODEL / 128);
    gemm_wmma_kernel<2><<<grid, 256, 0, stream>>>(ctx, WoT, bo, attn, NTOK, D_MODEL, D_MODEL);
  }

  // 5) x1 = LN(x + attn)  (f32 + f16 copies)
  add_ln_kernel<<<NTOK, 256, 0, stream>>>(x, attn, g1, be1, x1, x1h);

  // 6) FFN
  {
    dim3 grid1(NTOK / 64, D_FF / 128);
    gemm_wmma_kernel<1><<<grid1, 256, 0, stream>>>(x1h, W1T, b1, hbuf, NTOK, D_FF, D_MODEL);
    dim3 grid2(NTOK / 64, D_MODEL / 128);
    gemm_wmma_kernel<2><<<grid2, 256, 0, stream>>>(hbuf, W2T, b2, ff, NTOK, D_MODEL, D_FF);
  }

  // 7) out = LN(x1 + ff)
  add_ln_kernel<<<NTOK, 256, 0, stream>>>(x1, ff, g2, be2, out, ((_Float16*)nullptr));
}
